// DynaLoRALinear_42528766165276
// MI455X (gfx1250) — compile-verified, matrix operand order checked by reading, not run on hardware
//
#include <hip/hip_runtime.h>
#include <hip/hip_bf16.h>

// ---------------- problem dims (fixed by reference) ----------------
#define B_SZ   4
#define S_LEN  4096
#define D_MOD  2048
#define N_EXP  4
#define R_RANK 8

typedef __attribute__((ext_vector_type(16))) __bf16 v16bf;
typedef __attribute__((ext_vector_type(8)))  __bf16 v8bf;
typedef __attribute__((ext_vector_type(8)))  float  v8f;

// raw GCC-style 16B int vector — matches the async-builtin's parameter type
typedef int v4i_raw __attribute__((__vector_size__(16)));
typedef __attribute__((address_space(1))) v4i_raw* v4i_gptr;   // global (AS1)
typedef __attribute__((address_space(3))) v4i_raw* v4i_lptr;   // LDS (AS3)

// ---------------- workspace layout (bytes) ----------------
#define WS_G     0
#define WS_PART  256
#define WS_XBF   ((size_t)1 << 20)
#define WS_WEFF  (WS_XBF + (size_t)B_SZ * S_LEN * D_MOD * 2)

// ---------------- CDNA5 async copy (memory -> LDS, ASYNCcnt) ----------------
#if __has_builtin(__builtin_amdgcn_global_load_async_to_lds_b128)
#define ASYNC_COPY 1
#else
#define ASYNC_COPY 0
#endif

__device__ __forceinline__ void cp_async16(void* lds, const void* gmem) {
#if ASYNC_COPY
    // per-lane 16B: LDS[vdst_lane] = MEM[vaddr_lane]; tracked by ASYNCcnt.
    // Generic LDS ptr low 32 bits == LDS byte offset (ISA flat aperture rule),
    // so int->ptr casts produce correct AS1/AS3 operands without addrspacecast.
    __builtin_amdgcn_global_load_async_to_lds_b128(
        (v4i_gptr)(uintptr_t)gmem,
        (v4i_lptr)(uintptr_t)lds,
        /*imm offset*/ 0, /*cpol*/ 0);
#else
    *(v8bf*)lds = *(const v8bf*)gmem;     // sync fallback
#endif
}

template <int N>
__device__ __forceinline__ void wait_async() {
#if ASYNC_COPY
#if __has_builtin(__builtin_amdgcn_s_wait_asynccnt)
    __builtin_amdgcn_s_wait_asynccnt(N);
#else
    asm volatile("s_wait_asynccnt %0" :: "i"(N) : "memory");
#endif
#endif
}

// =====================================================================
// Kernel 1: stream x once -> bf16 copy + partial mean-pool sums.
// =====================================================================
__global__ void pool_convert_kernel(const float* __restrict__ x,
                                    float* __restrict__ part,
                                    __bf16* __restrict__ xbf) {
    const int d  = blockIdx.x * 256 + threadIdx.x;
    const int sc = blockIdx.y;
    const int b  = blockIdx.z;
    const size_t base = ((size_t)b * S_LEN + (size_t)sc * 256) * D_MOD + d;
    const float* xp = x + base;
    __bf16* xo = xbf + base;
    float sum = 0.f;
    #pragma unroll 8
    for (int s = 0; s < 256; ++s) {
        float v = xp[(size_t)s * D_MOD];
        sum += v;
        xo[(size_t)s * D_MOD] = (__bf16)v;
    }
    part[((size_t)b * 16 + sc) * D_MOD + d] = sum;
}

// =====================================================================
// Kernel 2: router logits + top-2 softmax gates. One small block.
// =====================================================================
__global__ void gate_kernel(const float* __restrict__ part,
                            const float* __restrict__ Wr,
                            float* __restrict__ g) {
    __shared__ float logits[B_SZ * N_EXP];
    const int t = threadIdx.x;
    if (t < B_SZ * N_EXP) {
        const int b = t / N_EXP, e = t % N_EXP;
        float acc = 0.f;
        for (int d = 0; d < D_MOD; ++d) {
            float p = 0.f;
            #pragma unroll
            for (int c = 0; c < 16; ++c)
                p += part[((size_t)b * 16 + c) * D_MOD + d];
            acc += p * Wr[(size_t)d * N_EXP + e];
        }
        logits[t] = acc * (1.0f / (float)S_LEN);
    }
    __syncthreads();
    if (t < B_SZ) {
        float l[N_EXP];
        #pragma unroll
        for (int e = 0; e < N_EXP; ++e) l[e] = logits[t * N_EXP + e];
        int i0 = 0;
        #pragma unroll
        for (int e = 1; e < N_EXP; ++e) if (l[e] > l[i0]) i0 = e;
        int i1 = -1;
        #pragma unroll
        for (int e = 0; e < N_EXP; ++e)
            if (e != i0 && (i1 < 0 || l[e] > l[i1])) i1 = e;
        const float e1 = __expf(l[i1] - l[i0]);
        const float inv = 1.f / (1.f + e1);
        float go[N_EXP] = {0.f, 0.f, 0.f, 0.f};
        go[i0] = inv;
        go[i1] = e1 * inv;
        #pragma unroll
        for (int e = 0; e < N_EXP; ++e) g[t * N_EXP + e] = go[e];
    }
}

// =====================================================================
// Kernel 3: W_eff[b][o][i] = W_base[o][i] + sum_e g[b,e] * (A[e] B[e])[i][o]
// =====================================================================
__global__ void weff_kernel(const float* __restrict__ Wb,
                            const float* __restrict__ A,
                            const float* __restrict__ Bm,
                            const float* __restrict__ g,
                            __bf16* __restrict__ weff) {
    const int b = blockIdx.y;
    const size_t idx = (size_t)blockIdx.x * 256 + threadIdx.x;
    const int o = (int)(idx / D_MOD);
    const int i = (int)(idx % D_MOD);
    float val = Wb[idx];
    #pragma unroll
    for (int e = 0; e < N_EXP; ++e) {
        const float ge = g[b * N_EXP + e];
        const float* Ap = A  + ((size_t)e * D_MOD + i) * R_RANK;
        const float* Bp = Bm + ((size_t)e * R_RANK) * D_MOD + o;
        float s = 0.f;
        #pragma unroll
        for (int r = 0; r < R_RANK; ++r)
            s += Ap[r] * Bp[(size_t)r * D_MOD];
        val += ge * s;
    }
    weff[(size_t)b * D_MOD * D_MOD + idx] = (__bf16)val;
}

// =====================================================================
// Kernel 4: per-batch NT GEMM  out[b] = xbf[b] (MxK) * weff[b]^T (NxK)
// 128x128x32 tile, 8 wave32 waves, wave tile 64x32 (4x2 WMMA accums).
// Double-buffered LDS fed by GLOBAL_LOAD_ASYNC_TO_LDS_B128 (ASYNCcnt).
// =====================================================================
#define BM 128
#define BN 128
#define BK 32
#define LDT 40   // padded LDS row stride (bf16 elems): 80B rows, bank-safe

__global__ __launch_bounds__(256, 1)
void gemm_kernel(const __bf16* __restrict__ xbf,
                 const __bf16* __restrict__ weff,
                 float* __restrict__ out) {
    __shared__ __align__(16) __bf16 As[2][BM * LDT];
    __shared__ __align__(16) __bf16 Bs[2][BN * LDT];

    const int b  = blockIdx.z;
    const int m0 = blockIdx.y * BM;
    const int n0 = blockIdx.x * BN;
    const __bf16* Ag = xbf  + (size_t)b * S_LEN * D_MOD;
    const __bf16* Bg = weff + (size_t)b * D_MOD * D_MOD;

    const int tid  = threadIdx.x;
    const int wid  = tid >> 5;
    const int lane = tid & 31;
    const int wm = wid >> 2;       // 0..1
    const int wn = wid & 3;        // 0..3
    const int lr = lane & 15;
    const int lh = lane >> 4;

    v8f acc[4][2];
    #pragma unroll
    for (int mi = 0; mi < 4; ++mi)
        #pragma unroll
        for (int ni = 0; ni < 2; ++ni)
            #pragma unroll
            for (int q = 0; q < 8; ++q) acc[mi][ni][q] = 0.f;

    // staging: thread t -> row t/2 (0..127), 32B chunk at (t&1)*16 elems
    const int srow   = tid >> 1;
    const int schunk = (tid & 1) * 16;

    // issue 4 async B128 copies (2 for A-tile, 2 for B-tile) for K-tile kt
    auto stage = [&](int kt, int p) {
        const __bf16* ga = Ag + (size_t)(m0 + srow) * D_MOD + kt + schunk;
        const __bf16* gb = Bg + (size_t)(n0 + srow) * D_MOD + kt + schunk;
        __bf16* la = &As[p][srow * LDT + schunk];
        __bf16* lb = &Bs[p][srow * LDT + schunk];
        cp_async16(la,     ga);
        cp_async16(la + 8, ga + 8);
        cp_async16(lb,     gb);
        cp_async16(lb + 8, gb + 8);
    };

    stage(0, 0);                    // prologue: tile 0 -> buffer 0
    int p = 0;
    for (int kt = 0; kt < D_MOD; kt += BK) {
        if (kt + BK < D_MOD) {
            stage(kt + BK, p ^ 1);  // next tile in flight during compute
            wait_async<4>();        // current tile's 4 copies landed
        } else {
            wait_async<0>();
        }
        __syncthreads();            // all waves' copies visible

        // fragment loads per ISA 16-bit layout:
        // lane<16: K 0..7 / 16..23 ; lane>=16: K 8..15 / 24..31
        v16bf afrag[4];
        #pragma unroll
        for (int mi = 0; mi < 4; ++mi) {
            const int row = wm * 64 + mi * 16 + lr;
            const v8bf lo = *(const v8bf*)(&As[p][row * LDT + lh * 8]);
            const v8bf hi = *(const v8bf*)(&As[p][row * LDT + lh * 8 + 16]);
            #pragma unroll
            for (int q = 0; q < 8; ++q) { afrag[mi][q] = lo[q]; afrag[mi][q + 8] = hi[q]; }
        }
        v16bf bfrag[2];
        #pragma unroll
        for (int ni = 0; ni < 2; ++ni) {
            const int col = wn * 32 + ni * 16 + lr;
            const v8bf lo = *(const v8bf*)(&Bs[p][col * LDT + lh * 8]);
            const v8bf hi = *(const v8bf*)(&Bs[p][col * LDT + lh * 8 + 16]);
            #pragma unroll
            for (int q = 0; q < 8; ++q) { bfrag[ni][q] = lo[q]; bfrag[ni][q + 8] = hi[q]; }
        }

        #pragma unroll
        for (int mi = 0; mi < 4; ++mi)
            #pragma unroll
            for (int ni = 0; ni < 2; ++ni)
                acc[mi][ni] = __builtin_amdgcn_wmma_f32_16x16x32_bf16(
                    false, afrag[mi], false, bfrag[ni],
                    (short)0, acc[mi][ni], false, false);

        __syncthreads();            // reads of buffer p done -> reusable
        p ^= 1;
    }

    // epilogue: C layout — VGPR r, lane L: M = r + (L>=16)*8, N = L%16
    float* op = out + ((size_t)b * S_LEN + m0) * D_MOD + n0;
    #pragma unroll
    for (int mi = 0; mi < 4; ++mi) {
        #pragma unroll
        for (int r = 0; r < 8; ++r) {
            const int row = wm * 64 + mi * 16 + r + lh * 8;
            #pragma unroll
            for (int ni = 0; ni < 2; ++ni) {
                const int col = wn * 32 + ni * 16 + lr;
                op[(size_t)row * D_MOD + col] = acc[mi][ni][r];
            }
        }
    }
}

// =====================================================================
extern "C" void kernel_launch(void* const* d_in, const int* in_sizes, int n_in,
                              void* d_out, int out_size, void* d_ws, size_t ws_size,
                              hipStream_t stream) {
    const float* x    = (const float*)d_in[0];   // [B,S,D]
    const float* Wb   = (const float*)d_in[1];   // [D,D]
    const float* A    = (const float*)d_in[2];   // [E,D,R]
    const float* Bm   = (const float*)d_in[3];   // [E,R,D]
    const float* Wr   = (const float*)d_in[4];   // [D,E]
    float* out = (float*)d_out;                  // [B,S,D] fp32

    char* ws = (char*)d_ws;
    float*  g    = (float*)(ws + WS_G);
    float*  part = (float*)(ws + WS_PART);
    __bf16* xbf  = (__bf16*)(ws + WS_XBF);
    __bf16* weff = (__bf16*)(ws + WS_WEFF);

    pool_convert_kernel<<<dim3(D_MOD / 256, 16, B_SZ), 256, 0, stream>>>(x, part, xbf);
    gate_kernel<<<1, 64, 0, stream>>>(part, Wr, g);
    weff_kernel<<<dim3((D_MOD * D_MOD) / 256, B_SZ), 256, 0, stream>>>(Wb, A, Bm, g, weff);
    gemm_kernel<<<dim3(D_MOD / BN, S_LEN / BM, B_SZ), 256, 0, stream>>>(xbf, weff, out);
}
